// DeepStructureAwareGATCoxModelV2_79448305041509
// MI455X (gfx1250) — matrix operand order repaired and checked by reading, not compiled, not run on hardware
//
#include <hip/hip_runtime.h>
#include <hip/hip_bf16.h>
#include <math.h>

typedef __attribute__((ext_vector_type(16))) _Float16 v16h;
typedef __attribute__((ext_vector_type(8)))  float    v8f;
typedef int v4i __attribute__((vector_size(16)));

#define G_     64
#define NPG_   128
#define N_     8192
#define E_     131072
#define FNODE_ 128
#define CLIN_  32
#define METAB_ 200
#define HID_   128
#define EH_    24
#define GD_    384
#define EPS_   1e-5f

#ifndef __has_builtin
#define __has_builtin(x) 0
#endif
#if __has_builtin(__builtin_amdgcn_global_load_async_to_lds_b128) && \
    __has_builtin(__builtin_amdgcn_s_wait_asynccnt)
#define ASYNC_STAGE 1
#else
#define ASYNC_STAGE 0
#endif

__device__ __forceinline__ float geluf(float x){ return 0.5f*x*(1.0f+erff(x*0.7071067811865475f)); }
__device__ __forceinline__ float leakyf(float x){ return x>0.f ? x : 0.2f*x; }

__device__ float atomicMaxF(float* addr, float val){
  int* ai = (int*)addr;
  int old = __float_as_int(*addr);
  while (__int_as_float(old) < val){
    int assumed = old;
    old = atomicCAS(ai, assumed, __float_as_int(val));
    if (old == assumed) break;
  }
  return __int_as_float(old);
}

// ---------------------------------------------------------------- utilities
__global__ void k_fill(float* p, float v, long long n){
  long long i = blockIdx.x*(long long)blockDim.x + threadIdx.x;
  if (i < n) p[i] = v;
}

__global__ void k_tohalf_pad(const float* __restrict__ src, _Float16* __restrict__ dst,
                             int R, int Csrc, int Cp){
  long long idx = blockIdx.x*(long long)blockDim.x + threadIdx.x;
  if (idx >= (long long)R*Cp) return;
  int r = (int)(idx / Cp), c = (int)(idx % Cp);
  dst[idx] = (_Float16)(c < Csrc ? src[(size_t)r*Csrc + c] : 0.f);
}

// ------------------------------------------------- CDNA5 WMMA f16 GEMM core
// C[M,Nc] (f32) = A[M,Kp] (f16, row major) x W[Nc,Kp]^T (f16, row major)
// Block tile: 64 rows x 128 cols. 8 waves; wave = 4 M-tiles x 1 N-tile.
// The 64xKp A panel is staged in LDS once per block (async-to-LDS when the
// toolchain exposes the builtin), so A fragments come from ds_load and the
// B fragment is reused across 4 WMMAs.
// Requires: M%64==0, Nc%128==0, Kp%32==0, Kp<=512.
__global__ void k_wmma_gemm(const _Float16* __restrict__ A,
                            const _Float16* __restrict__ B,
                            float* __restrict__ C,
                            int M, int Nc, int Kp,
                            long long sA, long long sB, long long sC){
  __shared__ _Float16 As[64*512];     // 64 KB max (Kp <= 512)
  const _Float16* Ab = A + sA*(long long)blockIdx.z;
  const _Float16* Bb = B + sB*(long long)blockIdx.z;
  float*          Cb = C + sC*(long long)blockIdx.z;
  int wave = threadIdx.x >> 5;
  int lane = threadIdx.x & 31;
  int m0 = blockIdx.x*64;
  int nt = blockIdx.y*8 + wave;
  int row = lane & 15;
  int hi  = lane >> 4;

  // ---- cooperative stage of contiguous 64xKp A panel into LDS
  {
    const char* gsrc = (const char*)(Ab + (size_t)m0*Kp);
    int bytes = 64*Kp*2;
    for (int o = (int)threadIdx.x*16; o < bytes; o += (int)blockDim.x*16){
#if ASYNC_STAGE
      __builtin_amdgcn_global_load_async_to_lds_b128(
          (__attribute__((address_space(1))) v4i*)(gsrc + o),
          (__attribute__((address_space(3))) v4i*)((char*)As + o), 0, 0);
#else
      *(uint4*)((char*)As + o) = *(const uint4*)(gsrc + o);
#endif
    }
#if ASYNC_STAGE
    __builtin_amdgcn_s_wait_asynccnt(0);
#endif
    __syncthreads();
  }

  v8f acc[4] = {};
  const _Float16* brow = Bb + (size_t)(nt*16 + row)*Kp;
  for (int k0 = 0; k0 < Kp; k0 += 32){
    if (k0 + 32 < Kp)
      __builtin_prefetch((const void*)(brow + k0 + 32), 0, 1);
    // B 32x16 striping: lane = N column; lanes 0-15 K in {0..7,16..23}; 16-31 {8..15,24..31}
    v16h b;
    const _Float16* bp = brow + k0 + hi*8;
#pragma unroll
    for (int j = 0; j < 8; ++j){ b[j] = bp[j]; b[8+j] = bp[16+j]; }
#pragma unroll
    for (int mt = 0; mt < 4; ++mt){
      // A 16x32 striping from LDS: lanes 0-15 K in {0..7,16..23}; 16-31 {8..15,24..31}
      v16h a;
      const _Float16* ap = As + (size_t)(mt*16 + row)*Kp + k0 + hi*8;
#pragma unroll
      for (int j = 0; j < 8; ++j){ a[j] = ap[j]; a[8+j] = ap[16+j]; }
      acc[mt] = __builtin_amdgcn_wmma_f32_16x16x32_f16(false, a, false, b, (short)0,
                                                       acc[mt], false, false);
    }
  }
  // D layout: VGPR r -> M = r (lanes 0-15) or r+8 (lanes 16-31); N = lane&15
  int n = nt*16 + row;
#pragma unroll
  for (int mt = 0; mt < 4; ++mt){
    int mr = m0 + mt*16 + hi*8;
#pragma unroll
    for (int r2 = 0; r2 < 8; ++r2)
      Cb[(size_t)(mr + r2)*Nc + n] = acc[mt][r2];
  }
}

// ------------------------------------------------------- structure features
__global__ void k_edge_build(const long long* __restrict__ ei, const float* __restrict__ ew,
                             float* deg, float* wdeg, float* cnt, float* adj){
  int e = blockIdx.x*blockDim.x + threadIdx.x;
  if (e >= E_) return;
  int s = (int)ei[e], d = (int)ei[E_ + e];
  float w = ew[e];
  atomicAdd(deg + s, 1.f);
  atomicAdd(wdeg + s, w);
  atomicAdd(cnt + d, 1.f);
  int g = s / NPG_, ls = s % NPG_, ld = d % NPG_;
  if (ls != ld){
    size_t o = (size_t)g*NPG_*NPG_ + (size_t)ls*NPG_ + ld;
    atomicMaxF(adj + o, 1.f);
  }
}

__global__ void k_sym(float* adj, _Float16* adjh){
  int idx = blockIdx.x*blockDim.x + threadIdx.x;
  const int per = NPG_*NPG_;
  if (idx >= G_*per) return;
  int g = idx/per, r = idx%per, i = r/NPG_, j = r%NPG_;
  if (j < i) return;
  size_t a = (size_t)g*per + (size_t)i*NPG_ + j;
  size_t b = (size_t)g*per + (size_t)j*NPG_ + i;
  float v = fmaxf(adj[a], adj[b]);
  adj[a] = v; adj[b] = v;
  adjh[a] = (_Float16)v; adjh[b] = (_Float16)v;
}

__global__ void k_nodefeat(const float* __restrict__ adj, const float* __restrict__ a2,
                           const float* __restrict__ deg, const float* __restrict__ wdeg,
                           float* ns, float* scal){
  int n = blockIdx.x*blockDim.x + threadIdx.x;
  if (n >= N_) return;
  int g = n/NPG_, i = n%NPG_;
  const float* ar = adj + (size_t)g*NPG_*NPG_ + (size_t)i*NPG_;
  const float* br = a2  + (size_t)g*NPG_*NPG_ + (size_t)i*NPG_;
  float ldeg = 0.f, tri = 0.f;
  for (int j = 0; j < NPG_; ++j){ ldeg += ar[j]; tri += ar[j]*br[j]; }
  tri *= 0.5f;
  float denom = fmaxf(ldeg*(ldeg-1.f)*0.5f, 1.f);
  float clus = tri/denom;
  float bridge = ldeg*(1.f - clus)/(float)NPG_;
  ns[n*5+0] = deg[n];
  ns[n*5+1] = wdeg[n];
  ns[n*5+2] = clus;
  ns[n*5+3] = tri/(float)NPG_;
  ns[n*5+4] = bridge;
  atomicMaxF(scal+0, deg[n]);
  atomicMaxF(scal+1, wdeg[n]);
  atomicMaxF(scal+2, bridge);
}

__global__ void k_assembleA(const float* __restrict__ x, const float* __restrict__ ns,
                            const float* __restrict__ scal, _Float16* A){
  long long idx = blockIdx.x*(long long)blockDim.x + threadIdx.x;
  if (idx >= (long long)N_*160) return;
  int n = (int)(idx/160), c = (int)(idx%160);
  float v = 0.f;
  if (c < FNODE_) v = x[(size_t)n*FNODE_ + c];
  else if (c < 133){
    int f = c - FNODE_;
    float raw = ns[n*5+f];
    if      (f == 0) v = raw / fmaxf(scal[0], 1.f);
    else if (f == 1) v = raw / fmaxf(scal[1], 1.f);
    else if (f == 4) v = raw / fmaxf(scal[2], 1.f);
    else             v = raw;
  }
  A[idx] = (_Float16)v;
}

// ---------------------------------------------------- epilogues / small MLPs
// out = gelu(LN(in + bias (+ res + resb)))   (block = 128 thr per row, W<=512)
__global__ void k_addln(const float* __restrict__ in, const float* __restrict__ bias,
                        const float* __restrict__ res, const float* __restrict__ resb,
                        const float* __restrict__ gam, const float* __restrict__ bet,
                        float* out, int W){
  __shared__ float buf[512];
  __shared__ float red[128];
  int row = blockIdx.x, t = threadIdx.x;
  const float* ir = in + (size_t)row*W;
  float s = 0.f;
  for (int c = t; c < W; c += 128){
    float v = ir[c] + bias[c];
    if (res) v += res[(size_t)row*W + c] + resb[c];
    buf[c] = v; s += v;
  }
  red[t] = s; __syncthreads();
  for (int off = 64; off > 0; off >>= 1){ if (t < off) red[t] += red[t+off]; __syncthreads(); }
  float mu = red[0]/(float)W; __syncthreads();
  float s2 = 0.f;
  for (int c = t; c < W; c += 128){ float d = buf[c]-mu; s2 += d*d; }
  red[t] = s2; __syncthreads();
  for (int off = 64; off > 0; off >>= 1){ if (t < off) red[t] += red[t+off]; __syncthreads(); }
  float inv = rsqrtf(red[0]/(float)W + EPS_);
  float* orow = out + (size_t)row*W;
  for (int c = t; c < W; c += 128)
    orow[c] = geluf((buf[c]-mu)*inv*gam[c] + bet[c]);
}

template<int ACT>
__global__ void k_dense(const float* __restrict__ in, const float* __restrict__ w,
                        const float* __restrict__ bias, float* out,
                        int M, int K, int Nout){
  long long idx = blockIdx.x*(long long)blockDim.x + threadIdx.x;
  if (idx >= (long long)M*Nout) return;
  int m = (int)(idx/Nout), o = (int)(idx%Nout);
  const float* ir = in + (size_t)m*K;
  const float* wr = w  + (size_t)o*K;
  float s = bias ? bias[o] : 0.f;
  for (int k = 0; k < K; ++k) s += ir[k]*wr[k];
  if (ACT == 1) s = geluf(s);
  out[idx] = s;
}

// ------------------------------------------------------------- edge pipeline
__global__ void k_edge_embed(const float* __restrict__ ew,
                             const float* __restrict__ w1, const float* __restrict__ b1,
                             const float* __restrict__ w2, const float* __restrict__ b2,
                             float* ee){
  int e = blockIdx.x*blockDim.x + threadIdx.x;
  if (e >= E_) return;
  float b  = ew[e];
  float f0 = b, f1 = b*b, f2 = log1pf(fabsf(b));
  float t[EH_];
  for (int o = 0; o < EH_; ++o)
    t[o] = geluf(w1[o*3]*f0 + w1[o*3+1]*f1 + w1[o*3+2]*f2 + b1[o]);
  float* er = ee + (size_t)e*EH_;
  for (int o = 0; o < EH_; ++o){
    float s = b2[o];
    for (int k = 0; k < EH_; ++k) s += w2[o*EH_+k]*t[k];
    er[o] = geluf(s);
  }
}

__global__ void k_meane_acc(const long long* __restrict__ ei, const float* __restrict__ ee,
                            float* acc){
  long long idx = blockIdx.x*(long long)blockDim.x + threadIdx.x;
  if (idx >= (long long)E_*EH_) return;
  int e = (int)(idx/EH_), j = (int)(idx%EH_);
  int d = (int)ei[E_ + e];
  atomicAdd(acc + (size_t)d*EH_ + j, ee[idx]);
}

__global__ void k_meane_fin(const float* __restrict__ cnt, float* acc){
  long long idx = blockIdx.x*(long long)blockDim.x + threadIdx.x;
  if (idx >= (long long)N_*EH_) return;
  acc[idx] /= fmaxf(cnt[idx/EH_], 1.f);
}

// fold attention vector into weight: wv[hh,k] = sum_c lin[(hh*outd+c),k]*att[hh,c]
__global__ void k_prep_att(const float* __restrict__ lin, const float* __restrict__ att,
                           float* wv, int h_, int outd, int K){
  int idx = blockIdx.x*blockDim.x + threadIdx.x;
  if (idx >= h_*K) return;
  int hh = idx/K, k = idx%K;
  float s = 0.f;
  for (int c = 0; c < outd; ++c)
    s += lin[((size_t)(hh*outd + c))*K + k]*att[hh*outd + c];
  wv[idx] = s;
}

__global__ void k_ae(const float* __restrict__ ee, const float* __restrict__ me,
                     const float* __restrict__ ve, float* ae, int h_){
  long long idx = blockIdx.x*(long long)blockDim.x + threadIdx.x;
  const int R = E_ + N_;
  if (idx >= (long long)R*h_) return;
  int r = (int)(idx/h_), hh = (int)(idx%h_);
  const float* row = (r < E_) ? ee + (size_t)r*EH_ : me + (size_t)(r-E_)*EH_;
  const float* vr  = ve + hh*EH_;
  float s = 0.f;
  for (int k = 0; k < EH_; ++k) s += row[k]*vr[k];
  ae[idx] = s;
}

__global__ void k_alpha(const long long* __restrict__ ei, const float* __restrict__ asrc,
                        const float* __restrict__ adst, const float* __restrict__ ae,
                        float* alpha, float* mx, int h_){
  long long idx = blockIdx.x*(long long)blockDim.x + threadIdx.x;
  const int R = E_ + N_;
  if (idx >= (long long)R*h_) return;
  int r = (int)(idx/h_), hh = (int)(idx%h_);
  int sf = (r < E_) ? (int)ei[r]      : r - E_;
  int df = (r < E_) ? (int)ei[E_ + r] : r - E_;
  float al = leakyf(asrc[sf*h_+hh] + adst[df*h_+hh] + ae[idx]);
  alpha[idx] = al;
  atomicMaxF(mx + df*h_ + hh, al);
}

__global__ void k_expsum(const long long* __restrict__ ei, float* alpha,
                         const float* __restrict__ mx, float* den, int h_){
  long long idx = blockIdx.x*(long long)blockDim.x + threadIdx.x;
  const int R = E_ + N_;
  if (idx >= (long long)R*h_) return;
  int r = (int)(idx/h_), hh = (int)(idx%h_);
  int df = (r < E_) ? (int)ei[E_ + r] : r - E_;
  float ex = expf(alpha[idx] - mx[df*h_+hh]);
  alpha[idx] = ex;
  atomicAdd(den + df*h_ + hh, ex);
}

__global__ void k_scatter(const long long* __restrict__ ei, const float* __restrict__ alpha,
                          const float* __restrict__ den, const float* __restrict__ xl,
                          float* out, int h_, int outd){
  int r = blockIdx.x;
  int fd = h_*outd;
  __shared__ float wc[8];
  int sf = (r < E_) ? (int)ei[r]      : r - E_;
  int df = (r < E_) ? (int)ei[E_ + r] : r - E_;
  if ((int)threadIdx.x < h_)
    wc[threadIdx.x] = alpha[(size_t)r*h_ + threadIdx.x]/den[df*h_ + threadIdx.x];
  __syncthreads();
  const float* xr = xl + (size_t)sf*fd;
  float* orow = out + (size_t)df*fd;
  for (int f = threadIdx.x; f < fd; f += blockDim.x)
    atomicAdd(orow + f, xr[f]*wc[f/outd]);
}

// --------------------------------------------------------------- pooling/head
__global__ void k_pool(const float* __restrict__ h, const float* __restrict__ gate,
                       float* pooled, int fd){
  int g = blockIdx.x, t = threadIdx.x;          // 128 threads, one per node
  __shared__ float sg[128];
  __shared__ float red[128];
  sg[t] = gate[g*NPG_ + t];
  red[t] = sg[t]; __syncthreads();
  for (int off = 64; off > 0; off >>= 1){ if (t < off) red[t] = fmaxf(red[t], red[t+off]); __syncthreads(); }
  float gm = red[0]; __syncthreads();
  float e = expf(sg[t] - gm);
  red[t] = e; __syncthreads();
  for (int off = 64; off > 0; off >>= 1){ if (t < off) red[t] += red[t+off]; __syncthreads(); }
  float Z = red[0]; __syncthreads();
  sg[t] = e/Z; __syncthreads();
  const float* hb = h + (size_t)g*NPG_*fd;
  float* pr = pooled + (size_t)g*3*fd;
  for (int f = t; f < fd; f += 128){
    float sm = 0.f, mxv = -3e38f, at = 0.f;
    for (int i = 0; i < NPG_; ++i){
      float v = hb[(size_t)i*fd + f];
      sm += v; mxv = fmaxf(mxv, v); at += v*sg[i];
    }
    pr[f] = sm/(float)NPG_; pr[fd+f] = mxv; pr[2*fd+f] = at;
  }
}

__global__ void k_fused(const float* __restrict__ gemb, const float* __restrict__ clin,
                        const float* __restrict__ met, float* fused){
  const int W = GD_ + CLIN_ + METAB_;
  int idx = blockIdx.x*blockDim.x + threadIdx.x;
  if (idx >= G_*W) return;
  int g = idx/W, j = idx%W;
  float v;
  if      (j < GD_)       v = gemb[g*GD_ + j];
  else if (j < GD_+CLIN_) v = clin[g*CLIN_ + (j-GD_)];
  else                    v = met[g*METAB_ + (j-GD_-CLIN_)];
  fused[idx] = v;
}

__global__ void k_final(const float* __restrict__ z, const float* __restrict__ rw,
                        const float* __restrict__ rb, float* out){
  int g = blockIdx.x*blockDim.x + threadIdx.x;
  if (g >= G_) return;
  float s = rb[0];
  for (int k = 0; k < HID_; ++k) s += z[(size_t)g*HID_ + k]*rw[k];
  out[g] = s;
}

// =============================================================== launch glue
extern "C" void kernel_launch(void* const* d_in, const int* in_sizes, int n_in,
                              void* d_out, int out_size, void* d_ws, size_t ws_size,
                              hipStream_t stream){
  (void)in_sizes; (void)n_in; (void)out_size; (void)ws_size;
  const float*     x    = (const float*)d_in[0];
  const float*     ew   = (const float*)d_in[1];
  const float*     clin = (const float*)d_in[2];
  const float*     met  = (const float*)d_in[3];
  const long long* ei   = (const long long*)d_in[4];
  // params: jax tree order (dict keys alphabetical)
  const float* ee_b1 = (const float*)d_in[5];
  const float* ee_b2 = (const float*)d_in[6];
  const float* ee_w1 = (const float*)d_in[7];
  const float* ee_w2 = (const float*)d_in[8];
  const float* f1_b  = (const float*)d_in[9];
  const float* f1_bb = (const float*)d_in[10];
  const float* f1_g  = (const float*)d_in[11];
  const float* f1_w  = (const float*)d_in[12];
  const float* f2_b  = (const float*)d_in[13];
  const float* f2_bb = (const float*)d_in[14];
  const float* f2_g  = (const float*)d_in[15];
  const float* f2_w  = (const float*)d_in[16];
  const float* np_b  = (const float*)d_in[89];
  const float* np_bb = (const float*)d_in[90];
  const float* np_g  = (const float*)d_in[91];
  const float* np_w  = (const float*)d_in[92];
  const float* r_b   = (const float*)d_in[93];
  const float* r_w   = (const float*)d_in[94];

  size_t off = 0;
  auto alloc = [&](size_t bytes)->char*{
    char* p = (char*)d_ws + off;
    off = (off + bytes + 255) & ~(size_t)255;
    return p;
  };
  float*    adj   = (float*)alloc(sizeof(float)*G_*NPG_*NPG_);
  float*    a2    = (float*)alloc(sizeof(float)*G_*NPG_*NPG_);
  _Float16* adjh  = (_Float16*)alloc(sizeof(_Float16)*G_*NPG_*NPG_);
  float*    deg   = (float*)alloc(sizeof(float)*N_);
  float*    wdeg  = (float*)alloc(sizeof(float)*N_);
  float*    cnt   = (float*)alloc(sizeof(float)*N_);
  float*    scal  = (float*)alloc(256);
  float*    ns    = (float*)alloc(sizeof(float)*N_*5);
  _Float16* Ah    = (_Float16*)alloc(sizeof(_Float16)*N_*512);
  _Float16* Wh    = (_Float16*)alloc(sizeof(_Float16)*512*512);
  float*    hbuf  = (float*)alloc(sizeof(float)*N_*512);
  float*    xl    = (float*)alloc(sizeof(float)*N_*512);
  float*    msg   = (float*)alloc(sizeof(float)*N_*512);
  float*    res   = (float*)alloc(sizeof(float)*N_*512);
  float*    eeb   = (float*)alloc(sizeof(float)*E_*EH_);
  float*    me    = (float*)alloc(sizeof(float)*N_*EH_);
  float*    asrc  = (float*)alloc(sizeof(float)*N_*8);
  float*    adst  = (float*)alloc(sizeof(float)*N_*8);
  float*    ae    = (float*)alloc(sizeof(float)*(size_t)(E_+N_)*4);
  float*    alp   = (float*)alloc(sizeof(float)*(size_t)(E_+N_)*4);
  float*    mx    = (float*)alloc(sizeof(float)*N_*8);
  float*    den   = (float*)alloc(sizeof(float)*N_*8);
  float*    wsrc  = (float*)alloc(sizeof(float)*8*512);
  float*    wdst  = (float*)alloc(sizeof(float)*8*512);
  float*    ve    = (float*)alloc(sizeof(float)*8*EH_);
  float*    gate  = (float*)alloc(sizeof(float)*N_);
  float*    pooled= (float*)alloc(sizeof(float)*G_*3*512);
  float*    ppraw = (float*)alloc(sizeof(float)*G_*GD_);
  float*    gemb  = (float*)alloc(sizeof(float)*G_*GD_);
  float*    fused = (float*)alloc(sizeof(float)*G_*(GD_+CLIN_+METAB_));
  float*    raw1  = (float*)alloc(sizeof(float)*G_*256);
  float*    z1    = (float*)alloc(sizeof(float)*G_*256);
  float*    raw2  = (float*)alloc(sizeof(float)*G_*128);
  float*    z2    = (float*)alloc(sizeof(float)*G_*128);
  float*    gate1 = xl;   // xl dead by the time pooling runs

  auto NB = [](long long n, int b){ return dim3((unsigned)((n + b - 1)/b)); };

  // ---------------- phase A: structure features
  (void)hipMemsetAsync(adj,  0, sizeof(float)*G_*NPG_*NPG_, stream);
  (void)hipMemsetAsync(deg,  0, sizeof(float)*N_, stream);
  (void)hipMemsetAsync(wdeg, 0, sizeof(float)*N_, stream);
  (void)hipMemsetAsync(cnt,  0, sizeof(float)*N_, stream);
  (void)hipMemsetAsync(scal, 0, 256, stream);
  k_edge_build<<<NB(E_,256),256,0,stream>>>(ei, ew, deg, wdeg, cnt, adj);
  k_sym<<<NB((long long)G_*NPG_*NPG_,256),256,0,stream>>>(adj, adjh);
  // a2 = adj @ adj   (adj symmetric -> A @ A^T form works)
  k_wmma_gemm<<<dim3(NPG_/64, NPG_/128, G_),256,0,stream>>>(adjh, adjh, a2,
      NPG_, NPG_, NPG_, NPG_*NPG_, NPG_*NPG_, NPG_*NPG_);
  k_nodefeat<<<NB(N_,256),256,0,stream>>>(adj, a2, deg, wdeg, ns, scal);

  // ---------------- node projection (K = 133 padded to 160)
  k_assembleA<<<NB((long long)N_*160,256),256,0,stream>>>(x, ns, scal, Ah);
  k_tohalf_pad<<<NB((long long)HID_*160,256),256,0,stream>>>(np_w, Wh, HID_, 133, 160);
  k_wmma_gemm<<<dim3(N_/64, HID_/128, 1),256,0,stream>>>(Ah, Wh, xl, N_, HID_, 160, 0,0,0);
  k_addln<<<N_,128,0,stream>>>(xl, np_b, nullptr, nullptr, np_g, np_bb, hbuf, HID_);

  // ---------------- edge embedding + per-dst mean
  k_edge_embed<<<NB(E_,256),256,0,stream>>>(ew, ee_w1, ee_b1, ee_w2, ee_b2, eeb);
  (void)hipMemsetAsync(me, 0, sizeof(float)*N_*EH_, stream);
  k_meane_acc<<<NB((long long)E_*EH_,256),256,0,stream>>>(ei, eeb, me);
  k_meane_fin<<<NB((long long)N_*EH_,256),256,0,stream>>>(cnt, me);

  // ---------------- GAT layers
  for (int l = 0; l < 4; ++l){
    int ind  = (l == 0) ? 128 : 512;
    int outd = 128;
    int h_   = (l == 3) ? 1 : 4;
    int fd   = h_*outd;
    int B = 17 + 18*l;
    const float* att_dst  = (const float*)d_in[B+0];
    const float* att_edge = (const float*)d_in[B+1];
    const float* att_src  = (const float*)d_in[B+2];
    const float* bias     = (const float*)d_in[B+3];
    const float* g_b1     = (const float*)d_in[B+4];
    const float* g_b2     = (const float*)d_in[B+5];
    const float* g_w1     = (const float*)d_in[B+6];
    const float* g_w2     = (const float*)d_in[B+7];
    const float* lin_e    = (const float*)d_in[B+8];
    const float* lin_w    = (const float*)d_in[B+9];
    const float* ln_b     = (const float*)d_in[B+10];
    const float* ln_g     = (const float*)d_in[B+11];
    const float* pp_b     = (const float*)d_in[B+12];
    const float* pp_bb    = (const float*)d_in[B+13];
    const float* pp_g     = (const float*)d_in[B+14];
    const float* pp_w     = (const float*)d_in[B+15];
    const float* res_b    = (const float*)d_in[B+16];
    const float* res_w    = (const float*)d_in[B+17];

    // fold attention vectors into projection weights
    k_prep_att<<<NB(h_*ind,128),128,0,stream>>>(lin_w, att_src, wsrc, h_, outd, ind);
    k_prep_att<<<NB(h_*ind,128),128,0,stream>>>(lin_w, att_dst, wdst, h_, outd, ind);
    k_prep_att<<<NB(h_*EH_,128),128,0,stream>>>(lin_e, att_edge, ve, h_, outd, EH_);

    // xl = h @ lin_w^T (WMMA, LDS-staged A panel)
    k_tohalf_pad<<<NB((long long)N_*ind,256),256,0,stream>>>(hbuf, Ah, N_, ind, ind);
    k_tohalf_pad<<<NB((long long)fd*ind,256),256,0,stream>>>(lin_w, Wh, fd, ind, ind);
    k_wmma_gemm<<<dim3(N_/64, fd/128, 1),256,0,stream>>>(Ah, Wh, xl, N_, fd, ind, 0,0,0);

    // attention logits + segment softmax over dst (incl. self loops)
    k_dense<0><<<NB((long long)N_*h_,256),256,0,stream>>>(hbuf, wsrc, nullptr, asrc, N_, ind, h_);
    k_dense<0><<<NB((long long)N_*h_,256),256,0,stream>>>(hbuf, wdst, nullptr, adst, N_, ind, h_);
    k_ae<<<NB((long long)(E_+N_)*h_,256),256,0,stream>>>(eeb, me, ve, ae, h_);
    k_fill<<<NB((long long)N_*h_,256),256,0,stream>>>(mx, -3e38f, (long long)N_*h_);
    (void)hipMemsetAsync(den, 0, sizeof(float)*N_*h_, stream);
    k_alpha<<<NB((long long)(E_+N_)*h_,256),256,0,stream>>>(ei, asrc, adst, ae, alp, mx, h_);
    k_expsum<<<NB((long long)(E_+N_)*h_,256),256,0,stream>>>(ei, alp, mx, den, h_);

    // message scatter
    (void)hipMemsetAsync(msg, 0, sizeof(float)*(size_t)N_*fd, stream);
    k_scatter<<<E_+N_,256,0,stream>>>(ei, alp, den, xl, msg, h_, outd);

    // residual projection (WMMA, reuses Ah = f16(h))
    k_tohalf_pad<<<NB((long long)fd*ind,256),256,0,stream>>>(res_w, Wh, fd, ind, ind);
    k_wmma_gemm<<<dim3(N_/64, fd/128, 1),256,0,stream>>>(Ah, Wh, res, N_, fd, ind, 0,0,0);

    // h = gelu(LN(msg + bias + res + res_b))
    k_addln<<<N_,128,0,stream>>>(msg, bias, res, res_b, ln_g, ln_b, hbuf, fd);

    // pooling: only the last layer's pooled output is consumed by the head
    if (l == 3){
      int gh = 64;  // max(16, fd//2) with fd=128
      k_dense<1><<<NB((long long)N_*gh,256),256,0,stream>>>(hbuf, g_w1, g_b1, gate1, N_, fd, gh);
      k_dense<0><<<NB((long long)N_,256),256,0,stream>>>(gate1, g_w2, g_b2, gate, N_, gh, 1);
      k_pool<<<G_,128,0,stream>>>(hbuf, gate, pooled, fd);
      k_dense<0><<<NB((long long)G_*GD_,128),128,0,stream>>>(pooled, pp_w, nullptr, ppraw, G_, 3*fd, GD_);
      k_addln<<<G_,128,0,stream>>>(ppraw, pp_b, nullptr, nullptr, pp_g, pp_bb, gemb, GD_);
    }
  }

  // ---------------- fusion head
  k_fused<<<NB((long long)G_*(GD_+CLIN_+METAB_),256),256,0,stream>>>(gemb, clin, met, fused);
  k_dense<0><<<NB((long long)G_*256,128),128,0,stream>>>(fused, f1_w, nullptr, raw1, G_, GD_+CLIN_+METAB_, 256);
  k_addln<<<G_,128,0,stream>>>(raw1, f1_b, nullptr, nullptr, f1_g, f1_bb, z1, 256);
  k_dense<0><<<NB((long long)G_*128,128),128,0,stream>>>(z1, f2_w, nullptr, raw2, G_, 256, 128);
  k_addln<<<G_,128,0,stream>>>(raw2, f2_b, nullptr, nullptr, f2_g, f2_bb, z2, 128);
  k_final<<<NB(G_,64),64,0,stream>>>(z2, r_w, r_b, (float*)d_out);
}
